// GAT_63204738728335
// MI455X (gfx1250) — matrix operand order repaired and checked by reading, not compiled, not run on hardware
//
#include <hip/hip_runtime.h>
#include <hip/hip_bf16.h>

// ---------------------------------------------------------------------------
// GATv2 x3 + LayerNorm for MI455X (gfx1250, wave32).
// GEMMs use v_wmma_f32_16x16x32_bf16 with register-staged LDS double
// buffering; edge phase is wave-per-edge coalesced gather/scatter with f32
// atomics resolving in the 192MB L2.
// ---------------------------------------------------------------------------

#define NN     50000
#define EE     800000
#define ELOOP  (NN + EE)       // edges + self loops
#define NEG_SLOPE 0.2f
#define LN_EPS    1e-5f

typedef __bf16 bf16;
typedef __attribute__((ext_vector_type(16))) __bf16 v16bf;
typedef __attribute__((ext_vector_type(8)))  float  v8f;

struct alignas(16) B16v { unsigned a, b, c, d; };     // 16-byte POD chunk
struct alignas(16) F4   { float x, y, z, w; };

// ---- order-preserving float <-> uint encoding for atomicMax --------------
__device__ __forceinline__ unsigned encf(float f) {
    unsigned u = __float_as_uint(f);
    return (u & 0x80000000u) ? ~u : (u | 0x80000000u);
}
__device__ __forceinline__ float decf(unsigned k) {
    return (k & 0x80000000u) ? __uint_as_float(k & 0x7FFFFFFFu)
                             : __uint_as_float(~k);
}
#define ENC_NEG_INF 0x007FFFFFu   // encf(-inf)

// ---------------------------------------------------------------------------
// utility kernels
// ---------------------------------------------------------------------------
__global__ void init_u32(unsigned* __restrict__ p, int cnt, unsigned val) {
    int i = blockIdx.x * 256 + threadIdx.x;
    if (i < cnt) p[i] = val;
}

__global__ void conv_f32_to_bf16(const float* __restrict__ src,
                                 unsigned short* __restrict__ dst, int cnt) {
    int i = blockIdx.x * 256 + threadIdx.x;
    if (i < cnt) *((bf16*)&dst[i]) = (bf16)src[i];
}

// W[k][n] (f32, K=256) -> Wt[n][k] (bf16) so GEMM B-tiles are K-contiguous.
__global__ void transpose_w_bf16(const float* __restrict__ W,
                                 unsigned short* __restrict__ Wt, int Nout) {
    int i = blockIdx.x * 256 + threadIdx.x;
    if (i >= 256 * Nout) return;
    int k = i / Nout, n = i - k * Nout;
    *((bf16*)&Wt[n * 256 + k]) = (bf16)W[i];
}

// ---------------------------------------------------------------------------
// Tiled bf16 WMMA GEMM:  C[M][Nout] = A[M][256] * Bt[Nout][256]^T + bias
// Block = 256 threads = 8 waves; block tile 64x64; wave tile 16x32.
// Double-buffered LDS, register-staged global loads: the global b128 loads
// for tile k+1 are issued before the ds_load/wmma work of tile k, and the
// s_wait_loadcnt lands only at the ds_store of the staged registers.
// ---------------------------------------------------------------------------
#define LDSTR 40   // padded LDS row stride (halfwords): conflict-free b128 reads

__global__ __launch_bounds__(256)
void gemm_bf16_k256(const unsigned short* __restrict__ A,
                    const unsigned short* __restrict__ Bt,
                    const float* __restrict__ bias,
                    float* __restrict__ C, int M, int Nout)
{
    __shared__ unsigned short As[2][64 * LDSTR];
    __shared__ unsigned short Bs[2][64 * LDSTR];

    const int t    = threadIdx.x;
    const int lane = t & 31;
    const int wave = t >> 5;
    const int mw   = (wave & 3) * 16;      // wave M offset in tile
    const int nw   = (wave >> 2) * 32;     // wave N offset in tile
    const int mBase = blockIdx.y * 64;
    const int nBase = blockIdx.x * 64;
    const int r    = t >> 2;               // tile row this thread stages
    const int cq   = (t & 3) * 8;          // 8 bf16 chunk within row
    const int l15  = lane & 15;
    const int half = lane >> 4;

    v8f acc0 = {0.f,0.f,0.f,0.f,0.f,0.f,0.f,0.f};
    v8f acc1 = {0.f,0.f,0.f,0.f,0.f,0.f,0.f,0.f};

    int gm = mBase + r; if (gm >= M) gm = M - 1;   // clamp (stores are guarded)
    const int gn = nBase + r;                      // Nout is a multiple of 64
    const unsigned short* gA = A  + (size_t)gm * 256 + cq;
    const unsigned short* gB = Bt + (size_t)gn * 256 + cq;
    const int stg = r * LDSTR + cq;                // LDS staging offset

    // ---- prologue: stage tile k=0 into buffer 0 ----
    {
        B16v ra = *(const B16v*)gA;
        B16v rb = *(const B16v*)gB;
        *(B16v*)&As[0][stg] = ra;
        *(B16v*)&Bs[0][stg] = rb;
    }
    __syncthreads();

    #pragma unroll
    for (int kk = 0; kk < 8; ++kk) {
        const int buf = kk & 1;

        // issue next tile's global loads first (latency hidden by wmma below)
        B16v ra, rb;
        if (kk < 7) {
            ra = *(const B16v*)(gA + (kk + 1) * 32);
            rb = *(const B16v*)(gB + (kk + 1) * 32);
            if (kk < 6)  // warm L2->WGP$ for tile k+2 (global_prefetch_b8)
                __builtin_prefetch(gA + (kk + 2) * 32, 0, 1);
        }

        // A fragment (ISA 16-bit 16x32 layout): row = lane%16,
        // lanes 0-15 hold K {0..7,16..23}, lanes 16-31 hold K {8..15,24..31}
        union { v16bf v; B16v q[2]; } fa, fb0, fb1;
        const unsigned short* pa = &As[buf][(mw + l15) * LDSTR + half * 8];
        fa.q[0] = *(const B16v*)pa;
        fa.q[1] = *(const B16v*)(pa + 16);

        // B fragments: col = lane%16, lanes 0-15 K {0..7,16..23}, lanes 16-31 K {8..15,24..31}
        const unsigned short* pb0 = &Bs[buf][(nw + l15) * LDSTR + half * 8];
        fb0.q[0] = *(const B16v*)pb0;
        fb0.q[1] = *(const B16v*)(pb0 + 16);
        const unsigned short* pb1 = &Bs[buf][(nw + 16 + l15) * LDSTR + half * 8];
        fb1.q[0] = *(const B16v*)pb1;
        fb1.q[1] = *(const B16v*)(pb1 + 16);

        acc0 = __builtin_amdgcn_wmma_f32_16x16x32_bf16(
                   false, fa.v, false, fb0.v, (short)0, acc0, false, false);
        acc1 = __builtin_amdgcn_wmma_f32_16x16x32_bf16(
                   false, fa.v, false, fb1.v, (short)0, acc1, false, false);

        if (kk < 7) {
            // buffer buf^1 was last read in iteration kk-1; the barrier at the
            // end of that iteration orders those reads before these writes.
            *(B16v*)&As[buf ^ 1][stg] = ra;
            *(B16v*)&Bs[buf ^ 1][stg] = rb;
            __syncthreads();
        }
    }

    // C/D layout: VGPR r holds M = r + (lane/16)*8, N = lane%16
    const int gn0 = nBase + nw + l15;
    const int gn1 = gn0 + 16;
    const float b0 = bias[gn0], b1 = bias[gn1];
    #pragma unroll
    for (int rr = 0; rr < 8; ++rr) {
        int gr = mBase + mw + rr + half * 8;
        if (gr < M) {
            C[(size_t)gr * Nout + gn0] = acc0[rr] + b0;
            C[(size_t)gr * Nout + gn1] = acc1[rr] + b1;
        }
    }
}

// ---------------------------------------------------------------------------
// Edge pass 1: logit[e][h] = sum_c att[h][c]*lrelu(xl[src]+xr[dst]);
// per-dst running max via encoded atomicMax. One wave per edge.
// ---------------------------------------------------------------------------
__global__ __launch_bounds__(256)
void edge_logit_max(const int* __restrict__ ei,
                    const float* __restrict__ xl, const float* __restrict__ xr,
                    const float* __restrict__ att,
                    float* __restrict__ logit, unsigned* __restrict__ mbuf,
                    int H)
{
    const int wid  = (blockIdx.x * 256 + threadIdx.x) >> 5;
    const int lane = threadIdx.x & 31;
    if (wid >= ELOOP) return;
    const int src = (wid < EE) ? ei[wid]      : (wid - EE);
    const int dst = (wid < EE) ? ei[EE + wid] : (wid - EE);
    const int Wd  = H << 7;                 // H*128 channels

    const F4* pl = (const F4*)(xl + (size_t)src * Wd) + lane * H;
    const F4* pr = (const F4*)(xr + (size_t)dst * Wd) + lane * H;
    const F4* pa = (const F4*)att + lane * H;

    float s = 0.f;
    for (int j = 0; j < H; ++j) {
        F4 a = pl[j], b = pr[j], c = pa[j];
        float v;
        v = a.x + b.x; v = v > 0.f ? v : NEG_SLOPE * v; s += c.x * v;
        v = a.y + b.y; v = v > 0.f ? v : NEG_SLOPE * v; s += c.y * v;
        v = a.z + b.z; v = v > 0.f ? v : NEG_SLOPE * v; s += c.z * v;
        v = a.w + b.w; v = v > 0.f ? v : NEG_SLOPE * v; s += c.w * v;
    }
    if (H == 2) { for (int m = 8;  m; m >>= 1) s += __shfl_xor(s, m, 16); }
    else        { for (int m = 16; m; m >>= 1) s += __shfl_xor(s, m, 32); }

    if ((lane & (32 / H - 1)) == 0) {
        int h = (H == 2) ? (lane >> 4) : 0;
        logit[(size_t)wid * H + h] = s;
        atomicMax(&mbuf[(size_t)dst * H + h], encf(s));
    }
}

// ---------------------------------------------------------------------------
// Edge pass 2: ex = exp(logit - max[dst]); denom[dst] += ex (in place on logit)
// ---------------------------------------------------------------------------
__global__ void edge_exp_denom(const int* __restrict__ ei,
                               float* __restrict__ exb,
                               const unsigned* __restrict__ mbuf,
                               float* __restrict__ denom, int H)
{
    int idx = blockIdx.x * 256 + threadIdx.x;
    if (idx >= ELOOP * H) return;
    int e = idx / H, h = idx - e * H;
    int dst = (e < EE) ? ei[EE + e] : (e - EE);
    float lm = decf(mbuf[dst * H + h]);
    float ex = __expf(exb[idx] - lm);
    exb[idx] = ex;
    atomicAdd(&denom[dst * H + h], ex);
}

// ---------------------------------------------------------------------------
// Edge pass 3: out[dst] += alpha * xl[src].  One wave per edge, f32 atomics.
// ---------------------------------------------------------------------------
__global__ __launch_bounds__(256)
void edge_aggregate(const int* __restrict__ ei,
                    const float* __restrict__ xl,
                    const float* __restrict__ exb,
                    const float* __restrict__ denom,
                    float* __restrict__ out, int H)
{
    const int wid  = (blockIdx.x * 256 + threadIdx.x) >> 5;
    const int lane = threadIdx.x & 31;
    if (wid >= ELOOP) return;
    const int src = (wid < EE) ? ei[wid]      : (wid - EE);
    const int dst = (wid < EE) ? ei[EE + wid] : (wid - EE);
    const int Wd  = H << 7;

    int h = (H == 2) ? (lane >> 4) : 0;
    float alpha = exb[(size_t)wid * H + h] / denom[(size_t)dst * H + h];

    const F4* pl = (const F4*)(xl + (size_t)src * Wd) + lane * H;
    float* po = out + (size_t)dst * Wd + (size_t)lane * H * 4;
    for (int j = 0; j < H; ++j) {
        F4 v = pl[j];
        atomicAdd(po + j * 4 + 0, alpha * v.x);
        atomicAdd(po + j * 4 + 1, alpha * v.y);
        atomicAdd(po + j * 4 + 2, alpha * v.z);
        atomicAdd(po + j * 4 + 3, alpha * v.w);
    }
}

// ---------------------------------------------------------------------------
// bias + LayerNorm + ReLU; writes f32 (final layer) or bf16 (next GEMM input).
// One wave per node.
// ---------------------------------------------------------------------------
__global__ __launch_bounds__(256)
void ln_relu(const float* __restrict__ agg, const float* __restrict__ bias,
             const float* __restrict__ gamma, const float* __restrict__ beta,
             float* __restrict__ outF, unsigned short* __restrict__ outBF,
             int Wd, int isLast)
{
    int n    = (blockIdx.x * 256 + threadIdx.x) >> 5;
    int lane = threadIdx.x & 31;
    if (n >= NN) return;
    int per  = Wd >> 5;           // 8 (Wd=256) or 4 (Wd=128)
    int base = lane * per;

    float v[8];
    const float* p = agg + (size_t)n * Wd + base;
    float s = 0.f, ss = 0.f;
    for (int j = 0; j < per; ++j) {
        float t = p[j] + bias[base + j];
        v[j] = t; s += t; ss += t * t;
    }
    for (int m = 16; m; m >>= 1) {
        s  += __shfl_xor(s,  m, 32);
        ss += __shfl_xor(ss, m, 32);
    }
    float mu  = s / (float)Wd;
    float var = ss / (float)Wd - mu * mu;
    float rs  = rsqrtf(var + LN_EPS);
    for (int j = 0; j < per; ++j) {
        float y = (v[j] - mu) * rs * gamma[base + j] + beta[base + j];
        y = y > 0.f ? y : 0.f;
        if (isLast) outF[(size_t)n * Wd + base + j] = y;
        else        *((bf16*)&outBF[(size_t)n * 256 + base + j]) = (bf16)y;
    }
}

// ---------------------------------------------------------------------------
// host orchestration
// ---------------------------------------------------------------------------
extern "C" void kernel_launch(void* const* d_in, const int* in_sizes, int n_in,
                              void* d_out, int out_size, void* d_ws, size_t ws_size,
                              hipStream_t stream)
{
    (void)in_sizes; (void)n_in; (void)out_size; (void)ws_size;
    const float* x  = (const float*)d_in[0];
    const int*   ei = (const int*)d_in[1];

    // ---- workspace carve (≈136 MB) ----
    char* w = (char*)d_ws;
    auto carve = [&](size_t bytes) { void* p = (void*)w; w += (bytes + 255) & ~(size_t)255; return p; };
    unsigned short* xbf = (unsigned short*)carve((size_t)NN * 256 * 2); // bf16 layer input
    float*          xlb = (float*)carve((size_t)NN * 256 * 4);          // source transform
    float*          xrb = (float*)carve((size_t)NN * 256 * 4);          // target transform, reused as agg out
    unsigned short* wlt = (unsigned short*)carve((size_t)256 * 256 * 2);
    unsigned short* wrt = (unsigned short*)carve((size_t)256 * 256 * 2);
    float*          exb = (float*)carve((size_t)ELOOP * 2 * 4);         // logits -> exp
    unsigned*       mb  = (unsigned*)carve((size_t)NN * 2 * 4);         // encoded segment max
    float*          db  = (float*)carve((size_t)NN * 2 * 4);            // segment denom

    // x -> bf16 once
    {
        int cnt = NN * 256;
        conv_f32_to_bf16<<<(cnt + 255) / 256, 256, 0, stream>>>(x, xbf, cnt);
    }

    const int Mtiles  = (NN + 63) / 64;
    const int eblocks = (ELOOP + 7) / 8;     // 8 waves (edges) per 256-thread block

    for (int l = 0; l < 3; ++l) {
        const float* Wl    = (const float*)d_in[2 + l * 8 + 0];
        const float* bl    = (const float*)d_in[2 + l * 8 + 1];
        const float* Wr    = (const float*)d_in[2 + l * 8 + 2];
        const float* br    = (const float*)d_in[2 + l * 8 + 3];
        const float* att   = (const float*)d_in[2 + l * 8 + 4];
        const float* bias  = (const float*)d_in[2 + l * 8 + 5];
        const float* gamma = (const float*)d_in[2 + l * 8 + 6];
        const float* beta  = (const float*)d_in[2 + l * 8 + 7];

        const int H    = (l == 2) ? 1 : 2;
        const int Nout = H * 128;

        // weights -> [Nout][256] bf16
        int wc = 256 * Nout;
        transpose_w_bf16<<<(wc + 255) / 256, 256, 0, stream>>>(Wl, wlt, Nout);
        transpose_w_bf16<<<(wc + 255) / 256, 256, 0, stream>>>(Wr, wrt, Nout);

        // init segment max (-inf encoded) and denom (0)
        init_u32<<<(NN * H + 255) / 256, 256, 0, stream>>>(mb, NN * H, ENC_NEG_INF);
        init_u32<<<(NN * H + 255) / 256, 256, 0, stream>>>((unsigned*)db, NN * H, 0u);

        // dense transforms via WMMA
        dim3 gg(Nout / 64, Mtiles);
        gemm_bf16_k256<<<gg, 256, 0, stream>>>(xbf, wlt, bl, xlb, NN, Nout);
        gemm_bf16_k256<<<gg, 256, 0, stream>>>(xbf, wrt, br, xrb, NN, Nout);

        // attention softmax + aggregation
        edge_logit_max<<<eblocks, 256, 0, stream>>>(ei, xlb, xrb, att, exb, mb, H);
        init_u32<<<(NN * Nout + 255) / 256, 256, 0, stream>>>((unsigned*)xrb, NN * Nout, 0u);
        edge_exp_denom<<<(ELOOP * H + 255) / 256, 256, 0, stream>>>(ei, exb, mb, db, H);
        edge_aggregate<<<eblocks, 256, 0, stream>>>(ei, xlb, exb, db, xrb, H);

        // bias + LayerNorm + ReLU (emit bf16 for next layer, f32 for output)
        ln_relu<<<(NN + 7) / 8, 256, 0, stream>>>(xrb, bias, gamma, beta,
                                                  (float*)d_out, xbf, Nout,
                                                  (l == 2) ? 1 : 0);
    }
}
